// Block_15436112461977
// MI455X (gfx1250) — matrix operand order repaired and checked by reading, not compiled, not run on hardware
//
#include <hip/hip_runtime.h>
#include <hip/hip_bf16.h>

typedef __bf16 bf16;
typedef __attribute__((ext_vector_type(8)))  bf16  v8bf;
typedef __attribute__((ext_vector_type(16))) bf16  v16bf;
typedef __attribute__((ext_vector_type(8)))  float v8f;

constexpr int Dd    = 1024;   // model dim
constexpr int Hh    = 4096;   // hidden dim
constexpr int Ee    = 8;      // experts
constexpr int TOTAL = 16384;  // B*S tokens
constexpr int CAP   = 2048;   // expert capacity

constexpr int BM  = 128;
constexpr int BN  = 128;
constexpr int BK  = 64;       // two WMMA k-steps per LDS tile
constexpr int LDT = BK + 8;   // padded LDS K-stride (elements); 144 B rows
constexpr int SST = BN + 8;   // padded stage stride for the epilogue restage
constexpr unsigned TILE_BYTES = BM * LDT * 2;  // one A or B buffer

// ---------------------------------------------------------------- helpers
__device__ __forceinline__ v8f wmma_bf16(v16bf a, v16bf b, v8f c) {
  // v_wmma_f32_16x16x32_bf16: (neg_a, A, neg_b, B, c_mod, C, reuse_a, reuse_b)
  return __builtin_amdgcn_wmma_f32_16x16x32_bf16(false, a, false, b, (short)0, c,
                                                 false, false);
}

// A fragment: lanes 0-15 hold K{0..15}, lanes 16-31 hold K{16..31} (contiguous)
__device__ __forceinline__ v16bf ld_frag_a(const bf16* p) {
  v8bf lo = *(const v8bf*)p;
  v8bf hi = *(const v8bf*)(p + 8);
  return __builtin_shufflevector(lo, hi, 0, 1, 2, 3, 4, 5, 6, 7,
                                 8, 9, 10, 11, 12, 13, 14, 15);
}
// B fragment: lanes 0-15 hold K{0..15}, lanes 16-31 hold K{16..31} (contiguous)
__device__ __forceinline__ v16bf ld_frag_b(const bf16* p) {
  v8bf lo = *(const v8bf*)p;
  v8bf hi = *(const v8bf*)(p + 8);
  return __builtin_shufflevector(lo, hi, 0, 1, 2, 3, 4, 5, 6, 7,
                                 8, 9, 10, 11, 12, 13, 14, 15);
}

__device__ __forceinline__ float gelu_tanh(float x) {
  const float c0 = 0.7978845608028654f;  // sqrt(2/pi)
  const float c1 = 0.044715f;
  return 0.5f * x * (1.0f + tanhf(c0 * (x + c1 * x * x * x)));
}

// async copy 16B/lane from global to LDS; tracked by ASYNCcnt
__device__ __forceinline__ void async_b128(unsigned lds_off, const bf16* gaddr) {
  asm volatile("global_load_async_to_lds_b128 %0, %1, off"
               :: "v"(lds_off), "v"(gaddr)
               : "memory");
}
__device__ __forceinline__ void wait_async_le8() {
  asm volatile("s_wait_asynccnt 0x8" ::: "memory");
}
__device__ __forceinline__ void wait_async_0() {
  asm volatile("s_wait_asynccnt 0x0" ::: "memory");
}

// ---------------------------------------------------------------- elementwise f32 -> bf16
__global__ __launch_bounds__(256) void cvt_bf16_kernel(
    const float* __restrict__ in, __hip_bfloat16* __restrict__ out, int n4) {
  const int i = blockIdx.x * 256 + threadIdx.x;
  if (i >= n4) return;
  const float4 v = ((const float4*)in)[i];
  out[i * 4 + 0] = __float2bfloat16(v.x);
  out[i * 4 + 1] = __float2bfloat16(v.y);
  out[i * 4 + 2] = __float2bfloat16(v.z);
  out[i * 4 + 3] = __float2bfloat16(v.w);
}

// ---------------------------------------------------------------- (batch,R,C) f32 -> (batch,C,R) bf16
__global__ __launch_bounds__(256) void transpose_cvt(
    const float* __restrict__ in, __hip_bfloat16* __restrict__ out, int R, int C) {
  const size_t bofs = (size_t)blockIdx.z * R * C;
  in += bofs;
  out += bofs;
  __shared__ float t[32][33];
  const int r0 = blockIdx.y * 32;
  const int c0 = blockIdx.x * 32;
  for (int i = threadIdx.y; i < 32; i += 8)
    t[i][threadIdx.x] = in[(size_t)(r0 + i) * C + (c0 + threadIdx.x)];
  __syncthreads();
  for (int i = threadIdx.y; i < 32; i += 8)
    out[(size_t)(c0 + i) * R + (r0 + threadIdx.x)] = __float2bfloat16(t[threadIdx.x][i]);
}

// ---------------------------------------------------------------- router: logits + softmax
// one wave32 per token; writes probs transposed (E, TOTAL)
__global__ __launch_bounds__(256) void router_kernel(
    const float* __restrict__ x, const float* __restrict__ gw,
    float* __restrict__ probs) {
  const int lane = threadIdx.x & 31;
  const int wave = threadIdx.x >> 5;
  const int t = blockIdx.x * 8 + wave;
  const float* xr = x + (size_t)t * Dd;
  float acc[Ee];
#pragma unroll
  for (int e = 0; e < Ee; ++e) acc[e] = 0.f;
  for (int k = lane; k < Dd; k += 32) {
    const float xv = xr[k];
    const float4 g0 = *(const float4*)(gw + (size_t)k * Ee);
    const float4 g1 = *(const float4*)(gw + (size_t)k * Ee + 4);
    acc[0] += xv * g0.x; acc[1] += xv * g0.y; acc[2] += xv * g0.z; acc[3] += xv * g0.w;
    acc[4] += xv * g1.x; acc[5] += xv * g1.y; acc[6] += xv * g1.z; acc[7] += xv * g1.w;
  }
#pragma unroll
  for (int e = 0; e < Ee; ++e)
#pragma unroll
    for (int off = 16; off > 0; off >>= 1) acc[e] += __shfl_xor(acc[e], off, 32);
  if (lane == 0) {
    float mx = acc[0];
#pragma unroll
    for (int e = 1; e < Ee; ++e) mx = fmaxf(mx, acc[e]);
    float p[Ee];
    float s = 0.f;
#pragma unroll
    for (int e = 0; e < Ee; ++e) { p[e] = expf(acc[e] - mx); s += p[e]; }
    const float inv = 1.f / s;
#pragma unroll
    for (int e = 0; e < Ee; ++e) probs[(size_t)e * TOTAL + t] = p[e] * inv;
  }
}

// ---------------------------------------------------------------- exact top-CAP via radix select
// one block per expert; probs > 0 so uint order == float order
__global__ __launch_bounds__(1024) void topk_kernel(
    const float* __restrict__ probs, int* __restrict__ tok_idx,
    float* __restrict__ scores) {
  const int e = blockIdx.x;
  const int tid = threadIdx.x;
  const float* p = probs + (size_t)e * TOTAL;
  __shared__ int hist[256];
  __shared__ unsigned sh_prefix;
  __shared__ int sh_need;
  __shared__ int cnt_gt, cnt_eq;

  unsigned prefix = 0;
  int need = CAP;
  for (int level = 3; level >= 0; --level) {
    const int shift = level * 8;
    for (int i = tid; i < 256; i += 1024) hist[i] = 0;
    __syncthreads();
    const unsigned mask_hi = (level == 3) ? 0u : (0xFFFFFFFFu << (shift + 8));
    for (int i = tid; i < TOTAL; i += 1024) {
      const unsigned u = __float_as_uint(p[i]);
      if ((u & mask_hi) == prefix) atomicAdd(&hist[(u >> shift) & 255], 1);
    }
    __syncthreads();
    if (tid == 0) {
      int acc = 0;
      int b = 255;
      for (; b > 0; --b) {
        if (acc + hist[b] >= need) break;
        acc += hist[b];
      }
      sh_prefix = prefix | ((unsigned)b << shift);
      sh_need = need - acc;
    }
    __syncthreads();
    prefix = sh_prefix;
    need = sh_need;
    __syncthreads();
  }
  const unsigned T = prefix;    // bit pattern of the CAP-th largest
  const int K_gt = CAP - need;  // count strictly greater than T
  if (tid == 0) { cnt_gt = 0; cnt_eq = 0; }
  __syncthreads();
  for (int i = tid; i < TOTAL; i += 1024) {
    const float v = p[i];
    const unsigned u = __float_as_uint(v);
    if (u > T) {
      const int pos = atomicAdd(&cnt_gt, 1);
      tok_idx[e * CAP + pos] = i;
      scores[e * CAP + pos] = v;
    } else if (u == T) {
      const int pos = atomicAdd(&cnt_eq, 1);
      if (pos < need) {
        tok_idx[e * CAP + K_gt + pos] = i;
        scores[e * CAP + K_gt + pos] = v;
      }
    }
  }
}

// ---------------------------------------------------------------- fused MoE GEMM
// MODE 0: shared FFN1  h  = gelu(x @ wsT + b)         (bf16 out, restaged)
// MODE 1: expert FFN1  h  = gelu(x[tok] @ wkT[e] + b) (gather, bf16 out, restaged)
// MODE 2: shared FFN2  out = h @ wsvT + b             (f32 store)
// MODE 3: expert FFN2  out += score * (h @ wvT[e]+b)  (f32 atomic scatter-add)
// 128x128 tile, BK=64, double-buffered LDS fed by GLOBAL_LOAD_ASYNC_TO_LDS_B128.
template <int MODE>
__global__ __launch_bounds__(256) void moe_gemm(
    const __hip_bfloat16* __restrict__ A_,   // activations (rows, K)
    const __hip_bfloat16* __restrict__ BT_,  // weights, N-major (E?, N, K)
    const float* __restrict__ bias_,         // (E?, N)
    const float* __restrict__ scores,        // (E, CAP)      [MODE 3]
    const int* __restrict__ tok_idx,         // (E, CAP)      [MODE 1,3]
    __hip_bfloat16* __restrict__ HoutBf,     // (M, N)        [MODE 0,1]
    float* __restrict__ OutF,                // (TOTAL, N)    [MODE 2,3]
    int K, int N) {
  constexpr bool GATHER = (MODE == 1);
  constexpr bool EXPERT_W = (MODE == 1 || MODE == 3);
  constexpr bool FFN1 = (MODE <= 1);
  constexpr bool SCATTER = (MODE == 3);

  const bf16* A = (const bf16*)A_;
  const bf16* BT = (const bf16*)BT_;
  const float* bias = bias_;

  __shared__ bf16 As[2][BM][LDT];   // 36864 B; reused as the bf16 output stage
  __shared__ bf16 Bs[2][BN][LDT];
  __shared__ int rowTok[BM];
  __shared__ float rowScore[BM];

  const int tid = threadIdx.x;
  const int mBase = blockIdx.y * BM;
  const int nBase = blockIdx.x * BN;

  if (EXPERT_W) {
    const int e = blockIdx.y >> 4;  // CAP/BM == 16 M-blocks per expert
    const int lBase = (blockIdx.y & 15) * BM;
    BT += (size_t)e * N * K;
    bias += (size_t)e * N;
    for (int i = tid; i < BM; i += 256) {
      rowTok[i] = tok_idx[e * CAP + lBase + i];
      if (SCATTER) rowScore[i] = scores[e * CAP + lBase + i];
    }
    __syncthreads();
  }

  // global->LDS slots: 1024 x 16B chunks per matrix per tile -> 4 per thread
  const bf16* aPtr[4];
  const bf16* bPtr[4];
  unsigned aLds[4], bLds[4];
#pragma unroll
  for (int i = 0; i < 4; ++i) {
    const int s = tid + i * 256;
    const int r = s >> 3;            // 0..127
    const int c = (s & 7) * 8;       // element offset, 16B chunks
    const int ga = GATHER ? rowTok[r] : (mBase + r);
    aPtr[i] = A + (size_t)ga * K + c;
    bPtr[i] = BT + (size_t)(nBase + r) * K + c;
    aLds[i] = (unsigned)(size_t)&As[0][r][c];  // LDS offset = addr[31:0]
    bLds[i] = (unsigned)(size_t)&Bs[0][r][c];
  }

  auto issue_tile = [&](int kt, int buf) {
    const unsigned add = buf ? TILE_BYTES : 0u;
    const size_t kofs = (size_t)kt * BK;
#pragma unroll
    for (int i = 0; i < 4; ++i) {
      async_b128(aLds[i] + add, aPtr[i] + kofs);
      async_b128(bLds[i] + add, bPtr[i] + kofs);
    }
  };

  // wave -> 32x64 output macro-tile (2 x 4 WMMA frags); 4 waves on M, 2 on N
  const int lane = tid & 31;
  const int wave = tid >> 5;
  const int wm = wave & 3;
  const int wn = wave >> 2;
  const int lrow = lane & 15;
  const int half = lane >> 4;
  const int koffA = half * 16;
  const int koffB = half * 16;

  const v8f vzero = {0.f, 0.f, 0.f, 0.f, 0.f, 0.f, 0.f, 0.f};
  v8f c[2][4];
#pragma unroll
  for (int mi = 0; mi < 2; ++mi)
#pragma unroll
    for (int ni = 0; ni < 4; ++ni) c[mi][ni] = vzero;

  const int KT = K >> 6;
  issue_tile(0, 0);
  for (int kt = 0; kt < KT; ++kt) {
    const int buf = kt & 1;
    if (kt + 1 < KT) {
      issue_tile(kt + 1, buf ^ 1);  // prefetch next tile into other buffer
      wait_async_le8();             // in-order: <=8 outstanding => tile kt landed
    } else {
      wait_async_0();               // drain fully (epilogue reuses As as stage)
    }
    __syncthreads();  // all waves' portions of tile kt are in LDS
#pragma unroll
    for (int ks = 0; ks < 2; ++ks) {  // two 16x16x32 k-steps per tile
      const int kb = ks * 32;
      v16bf af[2], bfr[4];
      af[0] = ld_frag_a(&As[buf][wm * 32 + lrow][kb + koffA]);
      af[1] = ld_frag_a(&As[buf][wm * 32 + 16 + lrow][kb + koffA]);
#pragma unroll
      for (int ni = 0; ni < 4; ++ni)
        bfr[ni] = ld_frag_b(&Bs[buf][wn * 64 + ni * 16 + lrow][kb + koffB]);
#pragma unroll
      for (int mi = 0; mi < 2; ++mi)
#pragma unroll
        for (int ni = 0; ni < 4; ++ni)
          c[mi][ni] = wmma_bf16(af[mi], bfr[ni], c[mi][ni]);
    }
    __syncthreads();  // compute(kt) done before tile kt+2 overwrites this buffer
  }

  // epilogue; C/D layout: VGPR j -> m = j + 8*half, lane&15 -> n
  if (FFN1) {
    // bias + gelu + bf16, restaged through LDS for coalesced b128 stores
    bf16* stage = &As[0][0][0];  // BM x SST bf16 = 34816 B (fits in As)
#pragma unroll
    for (int mi = 0; mi < 2; ++mi) {
#pragma unroll
      for (int ni = 0; ni < 4; ++ni) {
        const int nloc = wn * 64 + ni * 16 + lrow;
        const float bn = bias[nBase + nloc];
#pragma unroll
        for (int j = 0; j < 8; ++j) {
          const int rloc = wm * 32 + mi * 16 + half * 8 + j;
          stage[rloc * SST + nloc] = (bf16)gelu_tanh(c[mi][ni][j] + bn);
        }
      }
    }
    __syncthreads();
    // 2048 x 16B chunks -> 8 per thread; each wave stores 2x256B contiguous runs
#pragma unroll
    for (int i = 0; i < 8; ++i) {
      const int s = tid + i * 256;
      const int r = s >> 4;
      const int cc = (s & 15) * 8;
      const uint4 d = *(const uint4*)&stage[r * SST + cc];
      *(uint4*)(HoutBf + (size_t)(mBase + r) * N + (nBase + cc)) = d;
    }
  } else {
#pragma unroll
    for (int mi = 0; mi < 2; ++mi) {
#pragma unroll
      for (int ni = 0; ni < 4; ++ni) {
        const int n = nBase + wn * 64 + ni * 16 + lrow;
        const float bn = bias[n];
#pragma unroll
        for (int j = 0; j < 8; ++j) {
          const int rloc = wm * 32 + mi * 16 + half * 8 + j;
          const float v = c[mi][ni][j] + bn;
          if (SCATTER) {
            atomicAdd(&OutF[(size_t)rowTok[rloc] * N + n], v * rowScore[rloc]);
          } else {
            OutF[(size_t)(mBase + rloc) * N + n] = v;
          }
        }
      }
    }
  }
}

// ---------------------------------------------------------------- launcher
extern "C" void kernel_launch(void* const* d_in, const int* in_sizes, int n_in,
                              void* d_out, int out_size, void* d_ws, size_t ws_size,
                              hipStream_t stream) {
  const float* x     = (const float*)d_in[0];
  const float* gatew = (const float*)d_in[1];
  const float* wk    = (const float*)d_in[2];
  const float* bk    = (const float*)d_in[3];
  const float* wv    = (const float*)d_in[4];
  const float* bv    = (const float*)d_in[5];
  const float* ws_k  = (const float*)d_in[6];
  const float* bs_k  = (const float*)d_in[7];
  const float* ws_v  = (const float*)d_in[8];
  const float* bs_v  = (const float*)d_in[9];
  float* out = (float*)d_out;

  char* w = (char*)d_ws;
  size_t off = 0;
  auto carve = [&](size_t bytes) -> void* {
    void* p = (void*)(w + off);
    off += (bytes + 255) & ~(size_t)255;
    return p;
  };
  __hip_bfloat16* x_bf = (__hip_bfloat16*)carve((size_t)TOTAL * Dd * 2);
  __hip_bfloat16* wkT  = (__hip_bfloat16*)carve((size_t)Ee * Hh * Dd * 2);
  __hip_bfloat16* wvT  = (__hip_bfloat16*)carve((size_t)Ee * Dd * Hh * 2);
  __hip_bfloat16* wskT = (__hip_bfloat16*)carve((size_t)Hh * Dd * 2);
  __hip_bfloat16* wsvT = (__hip_bfloat16*)carve((size_t)Dd * Hh * 2);
  float* probs         = (float*)carve((size_t)Ee * TOTAL * 4);
  int* tok_idx         = (int*)carve((size_t)Ee * CAP * 4);
  float* scores        = (float*)carve((size_t)Ee * CAP * 4);
  __hip_bfloat16* hbuf = (__hip_bfloat16*)carve((size_t)TOTAL * Hh * 2);

  // 1) activations -> bf16
  {
    const int n4 = TOTAL * Dd / 4;
    cvt_bf16_kernel<<<n4 / 256, 256, 0, stream>>>(x, x_bf, n4);
  }
  // 2) weights -> bf16, transposed to N-major so GEMM B-tiles are K-contiguous
  transpose_cvt<<<dim3(Hh / 32, Dd / 32, Ee), dim3(32, 8), 0, stream>>>(wk, wkT, Dd, Hh);
  transpose_cvt<<<dim3(Dd / 32, Hh / 32, Ee), dim3(32, 8), 0, stream>>>(wv, wvT, Hh, Dd);
  transpose_cvt<<<dim3(Hh / 32, Dd / 32, 1), dim3(32, 8), 0, stream>>>(ws_k, wskT, Dd, Hh);
  transpose_cvt<<<dim3(Dd / 32, Hh / 32, 1), dim3(32, 8), 0, stream>>>(ws_v, wsvT, Hh, Dd);
  // 3) router softmax -> probs (E, TOTAL)
  router_kernel<<<TOTAL / 8, 256, 0, stream>>>(x, gatew, probs);
  // 4) exact expert-choice top-CAP (radix select)
  topk_kernel<<<Ee, 1024, 0, stream>>>(probs, tok_idx, scores);
  // 5) shared FFN1: h = gelu(x @ ws_k + bs_k)
  moe_gemm<0><<<dim3(Hh / BN, TOTAL / BM), 256, 0, stream>>>(
      x_bf, wskT, bs_k, nullptr, nullptr, hbuf, nullptr, Dd, Hh);
  // 6) shared FFN2: out = h @ ws_v + bs_v   (writes every output element)
  moe_gemm<2><<<dim3(Dd / BN, TOTAL / BM), 256, 0, stream>>>(
      hbuf, wsvT, bs_v, nullptr, nullptr, nullptr, out, Hh, Dd);
  // 7) expert FFN1 (fused gather): h = gelu(x[tok_idx] @ wk[e] + bk[e])
  moe_gemm<1><<<dim3(Hh / BN, TOTAL / BM), 256, 0, stream>>>(
      x_bf, wkT, bk, nullptr, tok_idx, hbuf, nullptr, Dd, Hh);
  // 8) expert FFN2: out += scores * (h @ wv[e] + bv[e])  (atomic scatter-add)
  moe_gemm<3><<<dim3(Dd / BN, TOTAL / BM), 256, 0, stream>>>(
      hbuf, wvT, bv, scores, tok_idx, nullptr, out, Hh, Dd);

  (void)in_sizes; (void)n_in; (void)out_size; (void)ws_size;
}
